// EnhancedHyperGeometricMemory_40295383171348
// MI455X (gfx1250) — matrix operand order repaired and checked by reading, not compiled, not run on hardware
//
#include <hip/hip_runtime.h>
#include <hip/hip_bf16.h>
#include <math.h>

// ---------------- problem constants ----------------
#define NTOK   8192      // B*S
#define INDIM  768
#define TD     72        // 3*D
#define DDIM   24
#define MSLOT  8192
#define TOPK   32
#define HDIM   256
#define QDIM   8

typedef __attribute__((ext_vector_type(16))) _Float16 v16h;
typedef __attribute__((ext_vector_type(8)))  _Float16 v8h;
typedef __attribute__((ext_vector_type(4)))  _Float16 v4h;
typedef __attribute__((ext_vector_type(8)))  float    v8f;

// =====================================================================
// WMMA GEMM (f16 operands, f32 accumulate):
//   C[M x N] = alpha * A[M x K] * B[N x K]^T + beta*C (+bias)
// A row-major (lda halves), B row-major N x K (ldb halves): K-contiguous.
// Each wave computes a 16(M) x 64(N) strip: one A fragment feeds 4 WMMAs.
// All 5 fragments are staged into distinct registers before any WMMA so
// the loads clause together and waits stay partial (load/WMMA overlap).
// 4 waves per block cover 4 adjacent M-tiles. M must be a multiple of 64.
// Layouts per CDNA5 ISA 7.12.2 (wave32).
// =====================================================================
__global__ __launch_bounds__(128) void wmma_gemm_f16(
    const _Float16* __restrict__ A, int lda,
    const _Float16* __restrict__ B, int ldb,
    float* __restrict__ C, int ldc,
    int Ndim, int Kdim,
    float alpha, float beta,
    const float* __restrict__ bias)
{
    const int wave  = threadIdx.x >> 5;
    const int lane  = threadIdx.x & 31;
    const int tileM = blockIdx.x * 4 + wave;
    const int n0    = blockIdx.y * 64;

    const int lh    = (lane < 16) ? 0 : 1;  // lane-half select
    const int kAoff = lh * 8;               // A: lanes 16-31 hold K+8 / K+24
    const int kBoff = lh * 8;               // B: lanes 16-31 hold K+8 / K+24

    const int m = tileM * 16 + (lane & 15);
    const _Float16* __restrict__ Arow = A + (size_t)m * lda;

    int ncol[4]; bool nval[4]; const _Float16* Brow[4];
#pragma unroll
    for (int j = 0; j < 4; ++j) {
        const int n = n0 + j * 16 + (lane & 15);
        ncol[j] = n;
        nval[j] = (n < Ndim);
        const int nc = nval[j] ? n : (Ndim - 1);   // clamp: safe garbage, never stored
        Brow[j] = B + (size_t)nc * ldb;
    }

    v8f acc[4];
#pragma unroll
    for (int j = 0; j < 4; ++j)
#pragma unroll
        for (int r = 0; r < 8; ++r) acc[j][r] = 0.0f;

    const int kmain = Kdim & ~31;
    for (int k0 = 0; k0 < kmain; k0 += 32) {
        __builtin_prefetch(Arow + k0 + 512, 0, 3);   // near-scope prefetch
        // ---- stage all fragments first (one load clause, partial waits) ----
        const v8h a0 = *(const v8h*)(Arow + k0 + kAoff);
        const v8h a1 = *(const v8h*)(Arow + k0 + 16 + kAoff);
        v8h b0[4], b1[4];
#pragma unroll
        for (int j = 0; j < 4; ++j) {
            b0[j] = *(const v8h*)(Brow[j] + k0 + kBoff);
            b1[j] = *(const v8h*)(Brow[j] + k0 + 16 + kBoff);
        }
        const v16h a = __builtin_shufflevector(a0, a1,
                           0, 1, 2, 3, 4, 5, 6, 7, 8, 9, 10, 11, 12, 13, 14, 15);
        // ---- then issue the 4 WMMAs ----
#pragma unroll
        for (int j = 0; j < 4; ++j) {
            const v16h b = __builtin_shufflevector(b0[j], b1[j],
                               0, 1, 2, 3, 4, 5, 6, 7, 8, 9, 10, 11, 12, 13, 14, 15);
            acc[j] = __builtin_amdgcn_wmma_f32_16x16x32_f16(
                         false, a, false, b, (short)0, acc[j], false, false);
        }
    }
    if (kmain < Kdim) {                      // guarded tail (K % 32 != 0)
        v16h a;
#pragma unroll
        for (int i = 0; i < 8; ++i) {
            const int k1 = kmain + kAoff + i;
            const int k2 = kmain + 16 + kAoff + i;
            a[i]     = (k1 < Kdim) ? Arow[k1] : (_Float16)0.0f;
            a[i + 8] = (k2 < Kdim) ? Arow[k2] : (_Float16)0.0f;
        }
#pragma unroll
        for (int j = 0; j < 4; ++j) {
            v16h b;
#pragma unroll
            for (int i = 0; i < 16; ++i) {
                const int kk = kmain + kBoff + (i < 8 ? i : i + 8);
                b[i] = (kk < Kdim) ? Brow[j][kk] : (_Float16)0.0f;
            }
            acc[j] = __builtin_amdgcn_wmma_f32_16x16x32_f16(
                         false, a, false, b, (short)0, acc[j], false, false);
        }
    }

#pragma unroll
    for (int j = 0; j < 4; ++j) {
        if (!nval[j]) continue;
#pragma unroll
        for (int r = 0; r < 8; ++r) {
            const int row = tileM * 16 + r + lh * 8;
            float* cp = C + (size_t)row * ldc + ncol[j];
            float v = alpha * acc[j][r];
            if (beta != 0.0f) v += beta * (*cp);
            if (bias)         v += bias[ncol[j]];
            *cp = v;
        }
    }
}

// packed f32 -> f16 conversion, 4 elements/thread
__global__ void cvtf16_kernel(const float* __restrict__ s,
                              _Float16* __restrict__ d, int n)
{
    const int i = (blockIdx.x * blockDim.x + threadIdx.x) * 4;
    if (i + 3 < n) {
        const float4 v = *(const float4*)(s + i);
        v4h h;
        h[0] = (_Float16)v.x; h[1] = (_Float16)v.y;
        h[2] = (_Float16)v.z; h[3] = (_Float16)v.w;
        *(v4h*)(d + i) = h;
    }
}

// =====================================================================
// DFT twiddle tables (exact via (k*n mod 256), stored f16 for WMMA B)
// + fractal scale scalar
// =====================================================================
__global__ void init_dft_kernel(_Float16* __restrict__ Cm, _Float16* __restrict__ Sm,
                                float* __restrict__ scale_out,
                                const float* __restrict__ fw)
{
    const int k = blockIdx.x, n = threadIdx.x;
    const int p = (k * n) & (HDIM - 1);
    const float ang = 6.283185307179586f * (float)p / (float)HDIM;
    Cm[k * HDIM + n] = (_Float16)cosf(ang);
    Sm[k * HDIM + n] = (_Float16)sinf(ang);
    if (k == 0 && n == 0) {
        float mx = fw[0];
        for (int s = 1; s < 4; ++s) mx = fmaxf(mx, fw[s]);
        float sum = 0.0f;
        for (int s = 0; s < 4; ++s) sum += expf(fw[s] - mx);
        float sc = 0.0f, inv2 = 1.0f;
        for (int s = 0; s < 4; ++s) { sc += expf(fw[s] - mx) / sum * inv2; inv2 *= 0.5f; }
        scale_out[0] = sc;
    }
}

// =====================================================================
// Post-encode: LN(72) -> GELU -> ricci einsum -> mean over triplet -> q[24]
// one wave32 per token
// =====================================================================
__global__ __launch_bounds__(32) void postenc_kernel(
    const float* __restrict__ ybuf, const float* __restrict__ g,
    const float* __restrict__ bb, const float* __restrict__ ricci,
    float* __restrict__ q)
{
    const int tok = blockIdx.x, t = threadIdx.x;
    const float* y = ybuf + (size_t)tok * TD;
    __shared__ float gl[TD];
    __shared__ float zbar[DDIM];

    float s = 0.0f;
    for (int j = t; j < TD; j += 32) s += y[j];
    for (int off = 16; off; off >>= 1) s += __shfl_xor(s, off, 32);
    const float mean = s / (float)TD;
    float v = 0.0f;
    for (int j = t; j < TD; j += 32) { const float d = y[j] - mean; v += d * d; }
    for (int off = 16; off; off >>= 1) v += __shfl_xor(v, off, 32);
    const float rstd = rsqrtf(v / (float)TD + 1e-5f);
    for (int j = t; j < TD; j += 32) {
        const float xn = (y[j] - mean) * rstd * g[j] + bb[j];
        gl[j] = 0.5f * xn * (1.0f + erff(xn * 0.7071067811865476f));
    }
    __syncthreads();
    if (t < DDIM) zbar[t] = (gl[3 * t] + gl[3 * t + 1] + gl[3 * t + 2]) * (1.0f / 3.0f);
    __syncthreads();
    if (t < DDIM) {
        float acc = 0.0f;
        for (int e = 0; e < DDIM; ++e) acc += zbar[e] * ricci[e * DDIM + t];
        q[(size_t)tok * DDIM + t] = acc;
    }
}

__global__ void knorm_kernel(const float* __restrict__ keys, float* __restrict__ kn)
{
    const int m = blockIdx.x * blockDim.x + threadIdx.x;
    if (m < MSLOT) {
        float s = 0.0f;
        for (int d = 0; d < DDIM; ++d) { const float v = keys[m * DDIM + d]; s += v * v; }
        kn[m] = s;
    }
}

// =====================================================================
// Fused cdist + top-32 + softmax weights + quantum gain, one block/token
// =====================================================================
__global__ __launch_bounds__(256) void topk_kernel(
    const float* __restrict__ q, const float* __restrict__ keys,
    const float* __restrict__ kn, const float* __restrict__ scalep,
    const float* __restrict__ qp,
    int* __restrict__ idxout, float* __restrict__ wout, float* __restrict__ gout)
{
    __shared__ float dist[MSLOT];
    __shared__ float qrow[DDIM];
    __shared__ float rv[256];
    __shared__ int   ri[256];
    __shared__ float selv[TOPK];
    __shared__ int   seli[TOPK];

    const int tok = blockIdx.x, t = threadIdx.x;
    if (t < DDIM) qrow[t] = q[(size_t)tok * DDIM + t];
    __syncthreads();
    float qq = 0.0f;
    for (int d = 0; d < DDIM; ++d) qq += qrow[d] * qrow[d];
    const float scale = scalep[0];

    for (int m = t; m < MSLOT; m += 256) {
        const float* kr = keys + (size_t)m * DDIM;
        float dot = 0.0f;
        for (int d = 0; d < DDIM; ++d) dot += qrow[d] * kr[d];
        const float d2 = qq + kn[m] - 2.0f * dot;
        dist[m] = sqrtf(fmaxf(d2, 0.0f)) * scale;
    }
    __syncthreads();

    for (int it = 0; it < TOPK; ++it) {
        float bv = 3.4e38f; int bi = 0x7fffffff;
        for (int m = t; m < MSLOT; m += 256) {
            const float v = dist[m];
            if (v < bv) { bv = v; bi = m; }
        }
        rv[t] = bv; ri[t] = bi;
        __syncthreads();
        for (int off = 128; off; off >>= 1) {
            if (t < off) {
                const float ov = rv[t + off]; const int oi = ri[t + off];
                if (ov < rv[t] || (ov == rv[t] && oi < ri[t])) { rv[t] = ov; ri[t] = oi; }
            }
            __syncthreads();
        }
        if (t == 0) { selv[it] = rv[0]; seli[it] = ri[0]; dist[ri[0]] = 3.4e38f; }
        __syncthreads();
    }

    if (t == 0) {
        const float mx = -selv[0];          // selv ascending -> max of (-d)
        float e[TOPK]; float sum = 0.0f;
        for (int k = 0; k < TOPK; ++k) { e[k] = expf(-selv[k] - mx); sum += e[k]; }
        float gacc = 0.0f;
        for (int k = 0; k < TOPK; ++k) {
            const float wk = e[k] / sum;
            wout[(size_t)tok * TOPK + k] = wk;
            idxout[(size_t)tok * TOPK + k] = seli[k];
            const float* qpr = qp + (size_t)seli[k] * QDIM;
            float ph = 0.0f;
            for (int qi = 0; qi < QDIM; ++qi) ph += tanhf(qpr[qi]);
            gacc += wk * ph * (1.0f / (float)QDIM);
        }
        gout[tok] = 1.0f + 0.02f * gacc;
    }
}

// phase = 2*pi*sigmoid(p) ; E = exp(i*phase)  (f16 output: feeds WMMA A)
__global__ void phase_kernel(const float* __restrict__ p,
                             _Float16* __restrict__ Er, _Float16* __restrict__ Ei)
{
    const int i = blockIdx.x * blockDim.x + threadIdx.x;
    const float ph = 6.283185307179586f / (1.0f + expf(-p[i]));
    Er[i] = (_Float16)cosf(ph);
    Ei[i] = (_Float16)sinf(ph);
}

// Kf *= exp(i*entangle_key[h])   (f32, in place)
__global__ void entangle_kernel(float* __restrict__ Kr, float* __restrict__ Ki,
                                const float* __restrict__ ek)
{
    const int i = blockIdx.x * blockDim.x + threadIdx.x;
    const int h = i & (HDIM - 1);
    const float c = cosf(ek[h]), s = sinf(ek[h]);
    const float kr = Kr[i], ki = Ki[i];
    Kr[i] = kr * c - ki * s;
    Ki[i] = ki * c + kr * s;
}

// Hmix[tok,h] = sum_k w_k * Hol[idx_k, h]
__global__ __launch_bounds__(256) void hmix_kernel(
    const int* __restrict__ idx, const float* __restrict__ w,
    const float* __restrict__ hre, const float* __restrict__ him,
    float* __restrict__ Hr, float* __restrict__ Hi)
{
    __shared__ int   si[TOPK];
    __shared__ float sw[TOPK];
    const int tok = blockIdx.x, t = threadIdx.x;
    if (t < TOPK) { si[t] = idx[(size_t)tok * TOPK + t]; sw[t] = w[(size_t)tok * TOPK + t]; }
    __syncthreads();
    float ar = 0.0f, ai = 0.0f;
    for (int k = 0; k < TOPK; ++k) {
        const size_t o = (size_t)si[k] * HDIM + t;
        ar += sw[k] * hre[o];
        ai += sw[k] * him[o];
    }
    Hr[(size_t)tok * HDIM + t] = ar;
    Hi[(size_t)tok * HDIM + t] = ai;
}

// G = conj(Kf) * Hmix   (f16 output: feeds WMMA A of the inverse DFT)
__global__ void gmul_kernel(const float* __restrict__ Kr, const float* __restrict__ Ki,
                            const float* __restrict__ Hr, const float* __restrict__ Hi,
                            _Float16* __restrict__ Gr, _Float16* __restrict__ Gi)
{
    const int i = blockIdx.x * blockDim.x + threadIdx.x;
    const float kr = Kr[i], ki = Ki[i], hr = Hr[i], hi = Hi[i];
    Gr[i] = (_Float16)(kr * hr + ki * hi);
    Gi[i] = (_Float16)(kr * hi - ki * hr);
}

// final: in-place LN(768) -> GELU -> * gain   (one block per token)
__global__ __launch_bounds__(256) void out_kernel(
    float* __restrict__ out, const float* __restrict__ g,
    const float* __restrict__ b, const float* __restrict__ gain)
{
    __shared__ float red[256];
    const int tok = blockIdx.x, t = threadIdx.x;
    float* row = out + (size_t)tok * INDIM;

    float s = 0.0f;
    for (int j = t; j < INDIM; j += 256) s += row[j];
    red[t] = s; __syncthreads();
    for (int off = 128; off; off >>= 1) { if (t < off) red[t] += red[t + off]; __syncthreads(); }
    const float mean = red[0] / (float)INDIM;
    __syncthreads();

    float v = 0.0f;
    for (int j = t; j < INDIM; j += 256) { const float d = row[j] - mean; v += d * d; }
    red[t] = v; __syncthreads();
    for (int off = 128; off; off >>= 1) { if (t < off) red[t] += red[t + off]; __syncthreads(); }
    const float rstd = rsqrtf(red[0] / (float)INDIM + 1e-5f);
    const float gn = gain[tok];

    for (int j = t; j < INDIM; j += 256) {
        const float xn = (row[j] - mean) * rstd * g[j] + b[j];
        row[j] = 0.5f * xn * (1.0f + erff(xn * 0.7071067811865476f)) * gn;
    }
}

// =====================================================================
extern "C" void kernel_launch(void* const* d_in, const int* in_sizes, int n_in,
                              void* d_out, int out_size, void* d_ws, size_t ws_size,
                              hipStream_t stream)
{
    (void)in_sizes; (void)n_in; (void)out_size; (void)ws_size;
    const float* x        = (const float*)d_in[0];
    const float* W_in     = (const float*)d_in[1];
    const float* b_in     = (const float*)d_in[2];
    const float* ln_in_g  = (const float*)d_in[3];
    const float* ln_in_b  = (const float*)d_in[4];
    const float* ricci    = (const float*)d_in[5];
    const float* fw       = (const float*)d_in[6];
    const float* keys     = (const float*)d_in[7];
    const float* kp_W     = (const float*)d_in[8];
    const float* kp_b     = (const float*)d_in[9];
    const float* ek       = (const float*)d_in[10];
    const float* holo_re  = (const float*)d_in[11];
    const float* holo_im  = (const float*)d_in[12];
    const float* ro_W     = (const float*)d_in[13];
    const float* ro_b     = (const float*)d_in[14];
    const float* W_out    = (const float*)d_in[15];
    const float* b_out    = (const float*)d_in[16];
    const float* ln_out_g = (const float*)d_in[17];
    const float* ln_out_b = (const float*)d_in[18];
    const float* qp       = (const float*)d_in[19];
    float* out = (float*)d_out;

    // ---- workspace carve-up (float slots; all chunks 16B-aligned) ----
    float* f = (float*)d_ws;
    size_t o = 0;
    float* q_ws  = f + o; o += (size_t)NTOK * DDIM;
    float* kn_ws = f + o; o += MSLOT;
    float* sc_ws = f + o; o += 16;
    int*   idx_ws = (int*)(f + o); o += (size_t)NTOK * TOPK;
    float* w_ws  = f + o; o += (size_t)NTOK * TOPK;
    float* gn_ws = f + o; o += NTOK;
    float* Kr    = f + o; o += (size_t)NTOK * HDIM;      // f32 DFT accum
    float* Ki    = f + o; o += (size_t)NTOK * HDIM;
    float* Hr    = f + o; o += (size_t)NTOK * HDIM;
    float* Hi    = f + o; o += (size_t)NTOK * HDIM;
    float* feat  = f + o; o += (size_t)NTOK * (2 * HDIM); // f32 IFFT accum
    float* tri   = f + o; o += (size_t)NTOK * TD;
    // f16 staging (2 halves per float slot)
    _Float16* xh   = (_Float16*)(f + o); o += (size_t)NTOK * INDIM / 2;
    _Float16* Erh  = (_Float16*)(f + o); o += (size_t)NTOK * HDIM / 2;
    _Float16* Eih  = (_Float16*)(f + o); o += (size_t)NTOK * HDIM / 2;
    _Float16* Cmh  = (_Float16*)(f + o); o += (size_t)HDIM * HDIM / 2;
    _Float16* Smh  = (_Float16*)(f + o); o += (size_t)HDIM * HDIM / 2;
    _Float16* Wih  = (_Float16*)(f + o); o += (size_t)TD * INDIM / 2;
    _Float16* kpWh = (_Float16*)(f + o); o += (size_t)HDIM * INDIM / 2;
    _Float16* roWh = (_Float16*)(f + o); o += (size_t)TD * (2 * HDIM) / 2;
    _Float16* Woh  = (_Float16*)(f + o); o += (size_t)INDIM * TD / 2;
    // aliases into dead regions
    float*    pbuf  = feat;               // dead before feat is written
    float*    ybuf  = tri;                // dead before tri is written
    _Float16* Grh   = Erh;                // E dead after forward DFT
    _Float16* Gih   = Eih;
    _Float16* feath = (_Float16*)Kr;      // Kr dead after gmul
    _Float16* trih  = (_Float16*)Ki;      // Ki dead after gmul

    auto gemm = [&](const _Float16* A, int lda, const _Float16* B, int ldb,
                    float* C, int ldc, int N_, int K_,
                    float alpha, float beta, const float* bias) {
        dim3 grid(NTOK / 64, (N_ + 63) / 64);   // 4 M-tiles per block (4 waves)
        wmma_gemm_f16<<<grid, 128, 0, stream>>>(A, lda, B, ldb, C, ldc,
                                                N_, K_, alpha, beta, bias);
    };
    auto cvt = [&](const float* s, _Float16* d, int n) {
        cvtf16_kernel<<<(n / 4 + 255) / 256, 256, 0, stream>>>(s, d, n);
    };

    // twiddles (f16) + fractal scale; f16 stage of x and all GEMM weights
    init_dft_kernel<<<HDIM, HDIM, 0, stream>>>(Cmh, Smh, sc_ws, fw);
    cvt(x,     xh,   NTOK * INDIM);
    cvt(W_in,  Wih,  TD * INDIM);
    cvt(kp_W,  kpWh, HDIM * INDIM);
    cvt(ro_W,  roWh, TD * 2 * HDIM);
    cvt(W_out, Woh,  INDIM * TD);

    // encode: y = x @ W_in^T + b_in ; then LN/GELU/ricci/mean -> q
    gemm(xh, INDIM, Wih, INDIM, ybuf, TD, TD, INDIM, 1.0f, 0.0f, b_in);
    postenc_kernel<<<NTOK, 32, 0, stream>>>(ybuf, ln_in_g, ln_in_b, ricci, q_ws);

    // distances + top-K + weights + gain
    knorm_kernel<<<MSLOT / 256, 256, 0, stream>>>(keys, kn_ws);
    topk_kernel<<<NTOK, 256, 0, stream>>>(q_ws, keys, kn_ws, sc_ws, qp,
                                          idx_ws, w_ws, gn_ws);

    // phase projection + E = exp(i*2pi*sigmoid(.))  (E stored f16)
    gemm(xh, INDIM, kpWh, INDIM, pbuf, HDIM, HDIM, INDIM, 1.0f, 0.0f, kp_b);
    phase_kernel<<<(NTOK * HDIM) / 256, 256, 0, stream>>>(pbuf, Erh, Eih);

    // forward DFT:  Kr = Er@C + Ei@S ;  Ki = Ei@C - Er@S   (f32 accum)
    gemm(Erh, HDIM, Cmh, HDIM, Kr, HDIM, HDIM, HDIM,  1.0f, 0.0f, nullptr);
    gemm(Eih, HDIM, Smh, HDIM, Kr, HDIM, HDIM, HDIM,  1.0f, 1.0f, nullptr);
    gemm(Eih, HDIM, Cmh, HDIM, Ki, HDIM, HDIM, HDIM,  1.0f, 0.0f, nullptr);
    gemm(Erh, HDIM, Smh, HDIM, Ki, HDIM, HDIM, HDIM, -1.0f, 1.0f, nullptr);

    entangle_kernel<<<(NTOK * HDIM) / 256, 256, 0, stream>>>(Kr, Ki, ek);
    hmix_kernel<<<NTOK, HDIM, 0, stream>>>(idx_ws, w_ws, holo_re, holo_im, Hr, Hi);
    gmul_kernel<<<(NTOK * HDIM) / 256, 256, 0, stream>>>(Kr, Ki, Hr, Hi, Grh, Gih);

    // inverse DFT -> feat = [vr | vi]  (vr = (Gr@C - Gi@S)/H ; vi = (Gi@C + Gr@S)/H)
    const float ih = 1.0f / (float)HDIM;
    gemm(Grh, HDIM, Cmh, HDIM, feat,        2 * HDIM, HDIM, HDIM,  ih, 0.0f, nullptr);
    gemm(Gih, HDIM, Smh, HDIM, feat,        2 * HDIM, HDIM, HDIM, -ih, 1.0f, nullptr);
    gemm(Gih, HDIM, Cmh, HDIM, feat + HDIM, 2 * HDIM, HDIM, HDIM,  ih, 0.0f, nullptr);
    gemm(Grh, HDIM, Smh, HDIM, feat + HDIM, 2 * HDIM, HDIM, HDIM,  ih, 1.0f, nullptr);

    // readout + output projection (f16 restage of activations)
    cvt(feat, feath, NTOK * 2 * HDIM);
    gemm(feath, 2 * HDIM, roWh, 2 * HDIM, tri, TD, TD, 2 * HDIM, 1.0f, 0.0f, ro_b);
    cvt(tri, trih, NTOK * TD);
    gemm(trih, TD, Woh, TD, out, INDIM, INDIM, TD, 1.0f, 0.0f, b_out);

    // final LN -> GELU -> * gain (in place on d_out)
    out_kernel<<<NTOK, 256, 0, stream>>>(out, ln_out_g, ln_out_b, gn_ws);
}
